// Transcoder_43293270343808
// MI455X (gfx1250) — matrix operand order repaired and checked by reading, not compiled, not run on hardware
//
#include <hip/hip_runtime.h>

// ---------------- problem constants ----------------
constexpr int BATCH  = 4096;
constexpr int D_IN   = 2048;
constexpr int D_OUT  = 2048;
constexpr int NFEAT  = 16384;
constexpr int TOPK   = 32;

// ---------------- types ----------------
typedef __attribute__((ext_vector_type(16))) __bf16   v16bf;
typedef __attribute__((ext_vector_type(8)))  float    v8f;
typedef __attribute__((ext_vector_type(8)))  unsigned short ushort8;

union BF16Frag {
    v16bf  v;
    ushort8 u[2];
};

__device__ __forceinline__ unsigned short f32_to_bf16(float f) {
    unsigned int u = __float_as_uint(f);
    unsigned int lsb = (u >> 16) & 1u;
    u += 0x7fffu + lsb;                 // round to nearest even
    return (unsigned short)(u >> 16);
}
__device__ __forceinline__ unsigned int pack_bf16x2(float a, float b) {
    return (unsigned int)f32_to_bf16(a) | ((unsigned int)f32_to_bf16(b) << 16);
}

// ---- CDNA5 async memory->LDS staging (ASYNCcnt path, bypasses VGPRs) ----
__device__ __forceinline__ void async_load_b128(unsigned lds_off, const void* gaddr) {
#if defined(__gfx1250__)
    asm volatile("global_load_async_to_lds_b128 %0, %1, off"
                 :: "v"(lds_off), "v"(gaddr)
                 : "memory");
#endif
}
__device__ __forceinline__ void wait_asynccnt0() {
#if defined(__gfx1250__)
    asm volatile("s_wait_asynccnt 0x0" ::: "memory");
#endif
}
__device__ __forceinline__ unsigned lds_offset_of(const void* p) {
    // generic (flat) pointer to a __shared__ object: low 32 bits = LDS byte offset
    return (unsigned)(size_t)p;
}

// ---------------- kernel 1: row norms + bf16 x_proc ----------------
__global__ void __launch_bounds__(256)
normalize_kernel(const float* __restrict__ x,
                 unsigned short* __restrict__ xb,
                 float* __restrict__ xnorm_out,
                 float* __restrict__ aux_out) {
    __shared__ float red[256];
    const int row = blockIdx.x;
    const int tid = threadIdx.x;
    const float* xr = x + (size_t)row * D_IN;

    float4 v0 = ((const float4*)xr)[tid * 2 + 0];
    float4 v1 = ((const float4*)xr)[tid * 2 + 1];
    float s = v0.x * v0.x + v0.y * v0.y + v0.z * v0.z + v0.w * v0.w
            + v1.x * v1.x + v1.y * v1.y + v1.z * v1.z + v1.w * v1.w;
    red[tid] = s;
    __syncthreads();
    for (int st = 128; st > 0; st >>= 1) {
        if (tid < st) red[tid] += red[tid + st];
        __syncthreads();
    }
    float norm = sqrtf(red[0]);
    if (tid == 0) xnorm_out[row] = norm;
    if (row == 0 && tid == 0) *aux_out = 0.0f;

    float inv = 1.0f / (norm + 1e-8f);
    uint4 p;
    p.x = pack_bf16x2(v0.x * inv, v0.y * inv);
    p.y = pack_bf16x2(v0.z * inv, v0.w * inv);
    p.z = pack_bf16x2(v1.x * inv, v1.y * inv);
    p.w = pack_bf16x2(v1.z * inv, v1.w * inv);
    ((uint4*)(xb + (size_t)row * D_IN))[tid] = p;
}

// ---------------- kernel 2: W_enc f32 -> bf16 ----------------
__global__ void __launch_bounds__(256)
convert_wenc_kernel(const float* __restrict__ W, unsigned short* __restrict__ Wb) {
    const size_t i = (size_t)blockIdx.x * blockDim.x + threadIdx.x;  // one uint4 (8 elems)
    float4 a = ((const float4*)W)[i * 2 + 0];
    float4 b = ((const float4*)W)[i * 2 + 1];
    uint4 p;
    p.x = pack_bf16x2(a.x, a.y);
    p.y = pack_bf16x2(a.z, a.w);
    p.z = pack_bf16x2(b.x, b.y);
    p.w = pack_bf16x2(b.z, b.w);
    ((uint4*)Wb)[i] = p;
}

// ---------------- kernel 3: W_dec (D_OUT x NFEAT) -> W_decT (NFEAT x D_OUT) ----------------
__global__ void __launch_bounds__(256)
transpose_wdec_kernel(const float* __restrict__ Wd, float* __restrict__ WdT) {
    __shared__ float tile[32][33];
    const int tx = threadIdx.x & 31;
    const int ty = threadIdx.x >> 5;     // 0..7
    const int bx = blockIdx.x;           // over NFEAT/32
    const int by = blockIdx.y;           // over D_OUT/32
#pragma unroll
    for (int i = 0; i < 4; ++i) {
        int r = ty + i * 8;
        tile[r][tx] = Wd[(size_t)(by * 32 + r) * NFEAT + bx * 32 + tx];
    }
    __syncthreads();
#pragma unroll
    for (int i = 0; i < 4; ++i) {
        int r = ty + i * 8;
        WdT[(size_t)(bx * 32 + r) * D_OUT + by * 32 + tx] = tile[tx][r];
    }
}

// ---------------- kernel 4: encoder GEMM (bf16 WMMA, f32 accum) ----------------
// pre[b, f] = sum_d xb[b,d] * Wb[f,d]  + b_enc[f]
constexpr int BM = 64, BN = 256, BK = 32;

__global__ void __launch_bounds__(256)
enc_gemm_kernel(const unsigned short* __restrict__ Xb,
                const unsigned short* __restrict__ Wb,
                const float* __restrict__ b_enc,
                float* __restrict__ pre) {
    __shared__ __attribute__((aligned(16))) unsigned short As[2][BM * BK];   //  8 KB
    __shared__ __attribute__((aligned(16))) unsigned short Bs[2][BN * BK];   // 32 KB

    const int tid  = threadIdx.x;
    const int lane = tid & 31;
    const int wave = tid >> 5;           // 0..7
    const int wm   = wave & 1;           // 2 M-subtiles of 32
    const int wn   = wave >> 1;          // 4 N-subtiles of 64
    const int mt   = lane & 15;
    const int kh   = lane >> 4;

    const int gm0 = blockIdx.y * BM;
    const int gn0 = blockIdx.x * BN;

    // staging: As 64x32 -> 1 x b128/thread; Bs 256x32 -> 4 x b128/thread (one row each)
    const int arow = tid >> 2, acol = (tid & 3) * 8;
    const unsigned short* gA = Xb + (size_t)(gm0 + arow) * D_IN + acol;
    const unsigned short* gB = Wb + (size_t)(gn0 + tid) * D_IN;

    v8f acc[2][4] = {};

    auto stage = [&](int buf, int k0) {
        async_load_b128(lds_offset_of(&As[buf][arow * BK + acol]), gA + k0);
#pragma unroll
        for (int c = 0; c < 4; ++c)
            async_load_b128(lds_offset_of(&Bs[buf][tid * BK + c * 8]), gB + k0 + c * 8);
    };

    stage(0, 0);
    const int KT = D_IN / BK;            // 64 k-steps
    for (int kt = 0; kt < KT; ++kt) {
        wait_asynccnt0();                // my async writes into buf (kt&1) done
        __syncthreads();                 // everyone's writes visible; prev reads done
        if (kt + 1 < KT) stage((kt + 1) & 1, (kt + 1) * BK);
        const int buf = kt & 1;

        // A fragments: lane holds K = kh*8..+7 and K = 16+kh*8..+7 of row (subtile + mt)
        BF16Frag a[2], b[4];
#pragma unroll
        for (int i = 0; i < 2; ++i) {
            const int base = (wm * 32 + i * 16 + mt) * BK + kh * 8;
            a[i].u[0] = *(const ushort8*)&As[buf][base];
            a[i].u[1] = *(const ushort8*)&As[buf][base + 16];
        }
        // B fragments: lane holds K = kh*8..+7 and K = 16+kh*8..+7 of column (subtile + mt)
#pragma unroll
        for (int j = 0; j < 4; ++j) {
            const int base = (wn * 64 + j * 16 + mt) * BK + kh * 8;
            b[j].u[0] = *(const ushort8*)&Bs[buf][base + 0];
            b[j].u[1] = *(const ushort8*)&Bs[buf][base + 16];
        }
#pragma unroll
        for (int i = 0; i < 2; ++i)
#pragma unroll
            for (int j = 0; j < 4; ++j)
                acc[i][j] = __builtin_amdgcn_wmma_f32_16x16x32_bf16(
                    false, a[i].v, false, b[j].v,
                    (short)0, acc[i][j], false, false);
    }

    // epilogue: D layout -> row = i*16 + 8*kh + r, col = j*16 + mt
#pragma unroll
    for (int j = 0; j < 4; ++j) {
        const int gn = gn0 + wn * 64 + j * 16 + mt;
        const float bias = b_enc[gn];
#pragma unroll
        for (int i = 0; i < 2; ++i) {
            const int rowbase = gm0 + wm * 32 + i * 16 + 8 * kh;
#pragma unroll
            for (int r = 0; r < 8; ++r)
                pre[(size_t)(rowbase + r) * NFEAT + gn] = acc[i][j][r] + bias;
        }
    }
}

// ---------------- kernel 5: per-row top-k threshold + relu scatter + compaction ----------------
__global__ void __launch_bounds__(256)
topk_scatter_kernel(float* __restrict__ hidden,        // pre_acts in, hidden_acts out
                    int* __restrict__ idxbuf,
                    float* __restrict__ valbuf,
                    int* __restrict__ counts) {
    __shared__ unsigned int red[256];
    __shared__ unsigned int cnt;
    const int row = blockIdx.x;
    const int tid = threadIdx.x;
    float* rowp = hidden + (size_t)row * NFEAT;

    // load my 64 elements as order-preserving uint keys (kept in registers)
    unsigned int keys[64];
#pragma unroll
    for (int t = 0; t < 64; ++t) {
        unsigned int bb = __float_as_uint(rowp[t * 256 + tid]);
        keys[t] = (bb & 0x80000000u) ? ~bb : (bb | 0x80000000u);
    }
    if (tid == 0) cnt = 0u;
    __syncthreads();

    // bitwise binary search for the TOPK-th largest key
    unsigned int prefix = 0u;
    for (int bit = 31; bit >= 0; --bit) {
        const unsigned int cand = prefix | (1u << bit);
        unsigned int local = 0u;
#pragma unroll
        for (int t = 0; t < 64; ++t) local += (keys[t] >= cand) ? 1u : 0u;
        red[tid] = local;
        __syncthreads();
        for (int s = 128; s > 0; s >>= 1) {
            if (tid < s) red[tid] += red[tid + s];
            __syncthreads();
        }
        if (red[0] >= (unsigned)TOPK) prefix = cand;
        __syncthreads();
    }

    // rewrite row: keep relu(value) at kept positions, zero elsewhere; compact pairs
#pragma unroll
    for (int t = 0; t < 64; ++t) {
        const int i = t * 256 + tid;
        const unsigned int kk = keys[t];
        const unsigned int bb = (kk & 0x80000000u) ? (kk & 0x7fffffffu) : ~kk;
        const float v = __uint_as_float(bb);
        float outv = 0.0f;
        if (kk >= prefix) {
            outv = v > 0.0f ? v : 0.0f;
            const unsigned int p = atomicAdd(&cnt, 1u);
            if (p < 64u) {
                idxbuf[row * 64 + p] = i;
                valbuf[row * 64 + p] = outv;
            }
        }
        rowp[i] = outv;
    }
    __syncthreads();
    if (tid == 0) counts[row] = (int)(cnt < 64u ? cnt : 64u);
}

// ---------------- kernel 6: sparse decode ----------------
__global__ void __launch_bounds__(256)
decode_kernel(const float* __restrict__ WdT,
              const float* __restrict__ b_dec,
              const int* __restrict__ idxbuf,
              const float* __restrict__ valbuf,
              const int* __restrict__ counts,
              const float* __restrict__ xnorm,
              float* __restrict__ y) {
    __shared__ int   sIdx[64];
    __shared__ float sVal[64];
    const int row = blockIdx.x;
    const int tid = threadIdx.x;
    const int count = counts[row];
    if (tid < 64 && tid < count) {
        sIdx[tid] = idxbuf[row * 64 + tid];
        sVal[tid] = valbuf[row * 64 + tid];
    }
    __syncthreads();

    float acc[8];
#pragma unroll
    for (int s = 0; s < 8; ++s) acc[s] = b_dec[s * 256 + tid];

    for (int c = 0; c < count; ++c) {
        const float* wr = WdT + (size_t)sIdx[c] * D_OUT;
        const float v = sVal[c];
#pragma unroll
        for (int s = 0; s < 8; ++s)
            acc[s] = fmaf(v, wr[s * 256 + tid], acc[s]);
    }
    const float xn = xnorm[row];
#pragma unroll
    for (int s = 0; s < 8; ++s)
        y[(size_t)row * D_OUT + s * 256 + tid] = acc[s] * xn;
}

// ---------------- launcher ----------------
extern "C" void kernel_launch(void* const* d_in, const int* in_sizes, int n_in,
                              void* d_out, int out_size, void* d_ws, size_t ws_size,
                              hipStream_t stream) {
    const float* x     = (const float*)d_in[0];
    const float* W_enc = (const float*)d_in[1];
    const float* b_enc = (const float*)d_in[2];
    const float* W_dec = (const float*)d_in[3];
    const float* b_dec = (const float*)d_in[4];

    float* out    = (float*)d_out;
    float* y_hat  = out;                                      // BATCH*D_OUT
    float* hidden = out + (size_t)BATCH * D_OUT;              // BATCH*NFEAT
    float* aux    = hidden + (size_t)BATCH * NFEAT;           // 1
    float* xnorm  = aux + 1;                                  // BATCH

    char* ws = (char*)d_ws;
    unsigned short* xb  = (unsigned short*)ws;                              // 16 MB
    unsigned short* Wb  = (unsigned short*)(ws + (size_t)16777216);         // 64 MB
    float*          WdT = (float*)(ws + (size_t)16777216 + 67108864);       // 128 MB
    char* p2 = ws + (size_t)16777216 + 67108864 + 134217728;
    int*   idxbuf = (int*)p2;                                               // 1 MB
    float* valbuf = (float*)(p2 + 1048576);                                 // 1 MB
    int*   counts = (int*)(p2 + 2097152);                                   // 16 KB

    normalize_kernel<<<BATCH, 256, 0, stream>>>(x, xb, xnorm, aux);
    convert_wenc_kernel<<<(NFEAT * (size_t)D_IN / 8) / 256, 256, 0, stream>>>(W_enc, Wb);
    transpose_wdec_kernel<<<dim3(NFEAT / 32, D_OUT / 32), 256, 0, stream>>>(W_dec, WdT);
    enc_gemm_kernel<<<dim3(NFEAT / BN, BATCH / BM), 256, 0, stream>>>(xb, Wb, b_enc, hidden);
    topk_scatter_kernel<<<BATCH, 256, 0, stream>>>(hidden, idxbuf, valbuf, counts);
    decode_kernel<<<BATCH, 256, 0, stream>>>(WdT, b_dec, idxbuf, valbuf, counts, xnorm, y_hat);
}
